// CrossModal_NN_4879082848546
// MI455X (gfx1250) — compile-verified, matrix operand order checked by reading, not run on hardware
//
#include <hip/hip_runtime.h>
#include <hip/hip_bf16.h>

// ---------------------------------------------------------------------------
// CDNA5 / gfx1250 implementation of CrossModal_NN forward.
// All matmuls run on v_wmma_f32_16x16x32_f16 (wave32 matrix cores).
// Bilinear kernel: c_va (A) fragments staged once in LDS and re-read
// just-in-time with ds_load_b128; W (B) fragments streamed from L2 in
// half-tile clauses behind sched_barriers.  ~100 live VGPRs -> no spills,
// high occupancy.
// ---------------------------------------------------------------------------

typedef _Float16 half_t;
typedef __attribute__((ext_vector_type(16))) _Float16 v16h;
typedef __attribute__((ext_vector_type(8)))  _Float16 v8h;
typedef __attribute__((ext_vector_type(8)))  float    v8f;

#define CB   1024    // batch
#define IMG  4096
#define AUD  1024
#define ENC  1024
#define SS   3
#define DD   256
#define SD   768     // S*D
#define BS   3072    // B*S
#define OUT1 512
#define LAT  128
#define NCLS 10
#define NORMC 0.0625f  // 1/sqrt(256)

// ---- fragment helpers ------------------------------------------------------
// ISA 16-bit A/B fragment (16x32): lane L -> row L%16, khalf = L/16.
// Lane reads 8 contiguous halfs at k = khalf*8 and 8 more at k = 16 + khalf*8.
__device__ __forceinline__ v16h load_frag16(const half_t* __restrict__ p) {
  union U { v16h v; v8h h[2]; } u;
  u.h[0] = *reinterpret_cast<const v8h*>(p);
  u.h[1] = *reinterpret_cast<const v8h*>(p + 16);
  return u.v;
}

// fragment re-read from an LDS register-image (32 contiguous bytes)
__device__ __forceinline__ v16h lds_frag(const half_t* p) {
  union U { v16h v; v8h h[2]; } u;
  u.h[0] = *reinterpret_cast<const v8h*>(p);
  u.h[1] = *reinterpret_cast<const v8h*>(p + 8);
  return u.v;
}

__device__ __forceinline__ v8f zero8() {
  v8f z;
#pragma unroll
  for (int i = 0; i < 8; ++i) z[i] = 0.0f;
  return z;
}

__device__ __forceinline__ float sigmoidf_(float x) {
  return 1.0f / (1.0f + __expf(-x));
}

// ---- elementwise kernels ---------------------------------------------------
__global__ void cvt_f32_f16_kernel(const float* __restrict__ in,
                                   half_t* __restrict__ out, int n) {
  int i = blockIdx.x * blockDim.x + threadIdx.x;
  if (i < n) out[i] = (half_t)in[i];
}

// Wt[o][jj][i] = W[o][i][jj]  (so bilinear B-fragment is K(=i)-contiguous)
__global__ void bil_transpose_kernel(const float* __restrict__ W,
                                     half_t* __restrict__ Wt, int n) {
  int g = blockIdx.x * blockDim.x + threadIdx.x;
  if (g >= n) return;
  int o  = g >> 16;
  int jj = (g >> 8) & 255;
  int i  = g & 255;
  Wt[g] = (half_t)W[(o << 16) + (i << 8) + jj];
}

// out = a*b, optional f32 and f16 outputs
__global__ void ew_mul_kernel(const float* __restrict__ a,
                              const float* __restrict__ b,
                              float* __restrict__ outF,
                              half_t* __restrict__ outH, int n) {
  int i = blockIdx.x * blockDim.x + threadIdx.x;
  if (i < n) {
    float v = a[i] * b[i];
    if (outF) outF[i] = v;
    if (outH) outH[i] = (half_t)v;
  }
}

// M = t_c * j * xv + (1-j) * xa
__global__ void combine_kernel(const float* __restrict__ j,
                               const float* __restrict__ xv,
                               const float* __restrict__ xa,
                               const float* __restrict__ tc,
                               float* __restrict__ out, int n) {
  int i = blockIdx.x * blockDim.x + threadIdx.x;
  if (i < n) {
    float jj = j[i];
    out[i] = tc[0] * jj * xv[i] + (1.0f - jj) * xa[i];
  }
}

// c_va = sigmoid((L4+L5)*alpha_v) -> f16 ; c_av = sigmoid((L6+L7)*alpha_a) -> f32
__global__ void gate_kernel(const float* __restrict__ L4, const float* __restrict__ L5,
                            const float* __restrict__ av,
                            const float* __restrict__ L6, const float* __restrict__ L7,
                            const float* __restrict__ aa,
                            half_t* __restrict__ cvaH, float* __restrict__ cavF, int n) {
  int i = blockIdx.x * blockDim.x + threadIdx.x;
  if (i < n) {
    cvaH[i] = (half_t)sigmoidf_((L4[i] + L5[i]) * av[i]);
    cavF[i] = sigmoidf_((L6[i] + L7[i]) * aa[i]);
  }
}

// ---- generic WMMA GEMM -----------------------------------------------------
// C[M,N] = act(A[M,K] @ Bt[N,K]^T + bias).  A,Bt f16 row-major (K contiguous).
// Block = 4 waves, block tile 64(M) x 64(N); wave w -> 16 rows x 64 cols.
// K-loop is software-pipelined (ping-pong fragment sets, handles odd K/32).
// Requires M % 64 == 0, K % 32 == 0; N arbitrary (clamped loads, guarded stores).
template <int ACT>
__global__ void wmma_gemm_kernel(const half_t* __restrict__ A,
                                 const half_t* __restrict__ Bt,
                                 const float* __restrict__ bias,
                                 float* __restrict__ Cf,
                                 half_t* __restrict__ Ch,
                                 int M, int N, int K) {
  const int lane  = threadIdx.x & 31;
  const int wave  = threadIdx.x >> 5;
  const int mlane = lane & 15;
  const int khalf = lane >> 4;
  const int rowBase = blockIdx.y * 64 + wave * 16;
  const int colBase = blockIdx.x * 64;

  v8f acc[4];
#pragma unroll
  for (int t = 0; t < 4; ++t) acc[t] = zero8();

  const half_t* arow = A + (size_t)(rowBase + mlane) * K + khalf * 8;
  const half_t* brow[4];
#pragma unroll
  for (int t = 0; t < 4; ++t) {
    int n = colBase + t * 16 + mlane;
    if (n >= N) n = N - 1;  // clamp; junk columns are never stored
    brow[t] = Bt + (size_t)n * K + khalf * 8;
  }

  const int niter = K >> 5;  // K/32 steps
  // stage 0 fragments
  v16h a0 = load_frag16(arow);
  v16h b0[4];
#pragma unroll
  for (int t = 0; t < 4; ++t) b0[t] = load_frag16(brow[t]);

  int k = 0;
#pragma unroll 1
  for (int i = 0; i + 2 <= niter; i += 2, k += 64) {
    // issue stage-1 loads (k+32) before consuming stage 0
    v16h a1 = load_frag16(arow + k + 32);
    v16h b1[4];
#pragma unroll
    for (int t = 0; t < 4; ++t) b1[t] = load_frag16(brow[t] + k + 32);
    __builtin_prefetch(arow + k + 128, 0, 0);  // global_prefetch_b8, speculative
    __builtin_amdgcn_sched_barrier(0);         // keep stage-1 loads above
#pragma unroll
    for (int t = 0; t < 4; ++t)
      acc[t] = __builtin_amdgcn_wmma_f32_16x16x32_f16(
          false, a0, false, b0[t], (short)0, acc[t], false, false);
    if (i + 2 < niter) {  // refill stage 0 with (k+64)
      a0 = load_frag16(arow + k + 64);
#pragma unroll
      for (int t = 0; t < 4; ++t) b0[t] = load_frag16(brow[t] + k + 64);
    }
    __builtin_amdgcn_sched_barrier(0);
#pragma unroll
    for (int t = 0; t < 4; ++t)
      acc[t] = __builtin_amdgcn_wmma_f32_16x16x32_f16(
          false, a1, false, b1[t], (short)0, acc[t], false, false);
  }
  if (niter & 1) {  // tail step (stage 0 already loaded)
#pragma unroll
    for (int t = 0; t < 4; ++t)
      acc[t] = __builtin_amdgcn_wmma_f32_16x16x32_f16(
          false, a0, false, b0[t], (short)0, acc[t], false, false);
  }

#pragma unroll
  for (int t = 0; t < 4; ++t) {
    int n = colBase + t * 16 + mlane;
    if (n < N) {
      float bv = bias ? bias[n] : 0.0f;
#pragma unroll
      for (int r = 0; r < 8; ++r) {
        int m = rowBase + r + 8 * khalf;
        float v = acc[t][r] + bv;
        if (ACT) v = fmaxf(v, 0.0f);
        if (Cf) Cf[(size_t)m * N + n] = v;
        if (Ch) Ch[(size_t)m * N + n] = (half_t)v;
      }
    }
  }
}

// ---- attention (S=3): 4 combos of softmax(q k^T)*NORM @ x ------------------
// One block (256 threads) per batch element.
__global__ void attention_kernel(const float* __restrict__ qv, const float* __restrict__ qa,
                                 const float* __restrict__ kv, const float* __restrict__ ka,
                                 const float* __restrict__ xv, const float* __restrict__ xa,
                                 float* aVAf, half_t* aVAh, float* aVf, half_t* aVh,
                                 float* aAVf, half_t* aAVh, float* aAf, half_t* aAh) {
  const int b = blockIdx.x;
  const int tid = threadIdx.x;
  __shared__ float sq[2][SS][DD];
  __shared__ float sk[2][SS][DD];
  __shared__ float sx[2][SS][DD];
  __shared__ float satt[4][SS][SS];

  const size_t base = (size_t)b * SD;
#pragma unroll
  for (int s = 0; s < SS; ++s) {
    sq[0][s][tid] = qv[base + s * DD + tid];
    sq[1][s][tid] = qa[base + s * DD + tid];
    sk[0][s][tid] = kv[base + s * DD + tid];
    sk[1][s][tid] = ka[base + s * DD + tid];
    sx[0][s][tid] = xv[base + s * DD + tid];
    sx[1][s][tid] = xa[base + s * DD + tid];
  }
  __syncthreads();

  // combo: 0 = (qv,ka,xa), 1 = (qv,kv,xv), 2 = (qa,kv,xv), 3 = (qa,ka,xa)
  if (tid < 36) {
    int c = tid / 9, p = tid % 9, s = p / 3, t = p % 3;
    int qs = (c < 2) ? 0 : 1;
    int ks = (c == 1 || c == 2) ? 0 : 1;
    float d = 0.0f;
    for (int i = 0; i < DD; ++i) d += sq[qs][s][i] * sk[ks][t][i];
    satt[c][s][t] = d;
  }
  __syncthreads();

  if (tid < 12) {
    int c = tid / 3, s = tid % 3;
    float a0 = satt[c][s][0], a1 = satt[c][s][1], a2 = satt[c][s][2];
    float mx = fmaxf(a0, fmaxf(a1, a2));
    float e0 = __expf(a0 - mx), e1 = __expf(a1 - mx), e2 = __expf(a2 - mx);
    float inv = NORMC / (e0 + e1 + e2);   // softmax first, then *1/sqrt(D)
    satt[c][s][0] = e0 * inv;
    satt[c][s][1] = e1 * inv;
    satt[c][s][2] = e2 * inv;
  }
  __syncthreads();

  float* outF[4] = {aVAf, aVf, aAVf, aAf};
  half_t* outH[4] = {aVAh, aVh, aAVh, aAh};
  const int d = tid;
#pragma unroll
  for (int c = 0; c < 4; ++c) {
    int xs = (c == 1 || c == 2) ? 0 : 1;
#pragma unroll
    for (int s = 0; s < SS; ++s) {
      float v = satt[c][s][0] * sx[xs][0][d] +
                satt[c][s][1] * sx[xs][1][d] +
                satt[c][s][2] * sx[xs][2][d];
      outF[c][base + s * DD + d] = v;
      outH[c][base + s * DD + d] = (half_t)v;
    }
  }
}

// ---- fused bilinear: j[bs,o] = sigmoid(c_va . (W[o] @ c_av)) ---------------
// One wave per (16-row tile, output channel o).  G = c_va @ Wt[o]^T with WMMA.
// A (c_va) fragments: loaded from global ONCE, staged in LDS (8 KB register
// image), re-read just-in-time with ds_load_b128 one fragment ahead of the
// consuming WMMA.  B (W[o]) fragments: streamed from L2 in two 8x b128 half-
// tile clauses per nt, fenced above their 4-WMMA chains by sched_barrier.
// Peak liveness ~100 VGPRs -> no spills + high occupancy.
__global__ void bilinear_kernel(const half_t* __restrict__ cvaH,
                                const float* __restrict__ cavF,
                                const half_t* __restrict__ Wt,  // [o][jj][i]
                                float* __restrict__ J) {
  const int lane = threadIdx.x;
  const int rowBase = blockIdx.x * 16;
  const int o = blockIdx.y;
  const int mlane = lane & 15;
  const int khalf = lane >> 4;

  __shared__ __align__(16) half_t sA[32 * 128];  // 8 KB fragment image
  __shared__ float red[32][8];

  // stage this row-tile's c_va fragments into LDS (per-lane register image)
  {
    const half_t* arow = cvaH + (size_t)(rowBase + mlane) * DD + khalf * 8;
#pragma unroll
    for (int kt = 0; kt < 8; ++kt) {
      union { v16h v; v8h h[2]; } u;
      u.v = load_frag16(arow + kt * 32);
      v8h* p = reinterpret_cast<v8h*>(sA + (size_t)lane * 128 + kt * 16);
      p[0] = u.h[0];
      p[1] = u.h[1];
    }
  }
  __syncthreads();
  const half_t* sAl = sA + (size_t)lane * 128;

  float part[8];
#pragma unroll
  for (int r = 0; r < 8; ++r) part[r] = 0.0f;

  const half_t* brow = Wt + (size_t)o * (DD * DD) + (size_t)mlane * DD + khalf * 8;
  // epilogue base: lane's column is n, lane's rows are rowBase + 8*khalf + r
  const float* cavBase = cavF + (size_t)(rowBase + 8 * khalf) * DD + mlane;

#pragma unroll 1
  for (int nt = 0; nt < 16; ++nt) {
    const half_t* bnt = brow + (size_t)nt * 16 * DD;
    v8f acc0 = zero8();
    v8f acc1 = zero8();
    float cav[8];
    // ---- half 0: kt = 0..3 ----
    {
      v16h b[4];
#pragma unroll
      for (int kt = 0; kt < 4; ++kt) b[kt] = load_frag16(bnt + kt * 32);
      const int nb = nt * 16;
#pragma unroll
      for (int r = 0; r < 8; ++r) cav[r] = cavBase[(size_t)r * DD + nb];
      __builtin_amdgcn_sched_barrier(0);  // keep the load clause above
      v16h a0 = lds_frag(sAl + 0 * 16);
      v16h a1 = lds_frag(sAl + 1 * 16);
      acc0 = __builtin_amdgcn_wmma_f32_16x16x32_f16(
          false, a0, false, b[0], (short)0, acc0, false, false);
      v16h a2 = lds_frag(sAl + 2 * 16);
      acc1 = __builtin_amdgcn_wmma_f32_16x16x32_f16(
          false, a1, false, b[1], (short)0, acc1, false, false);
      v16h a3 = lds_frag(sAl + 3 * 16);
      acc0 = __builtin_amdgcn_wmma_f32_16x16x32_f16(
          false, a2, false, b[2], (short)0, acc0, false, false);
      acc1 = __builtin_amdgcn_wmma_f32_16x16x32_f16(
          false, a3, false, b[3], (short)0, acc1, false, false);
    }
    // ---- half 1: kt = 4..7 ----
    {
      v16h b[4];
#pragma unroll
      for (int kt = 0; kt < 4; ++kt) b[kt] = load_frag16(bnt + (4 + kt) * 32);
      __builtin_amdgcn_sched_barrier(0);
      v16h a4 = lds_frag(sAl + 4 * 16);
      v16h a5 = lds_frag(sAl + 5 * 16);
      acc0 = __builtin_amdgcn_wmma_f32_16x16x32_f16(
          false, a4, false, b[0], (short)0, acc0, false, false);
      v16h a6 = lds_frag(sAl + 6 * 16);
      acc1 = __builtin_amdgcn_wmma_f32_16x16x32_f16(
          false, a5, false, b[1], (short)0, acc1, false, false);
      v16h a7 = lds_frag(sAl + 7 * 16);
      acc0 = __builtin_amdgcn_wmma_f32_16x16x32_f16(
          false, a6, false, b[2], (short)0, acc0, false, false);
      acc1 = __builtin_amdgcn_wmma_f32_16x16x32_f16(
          false, a7, false, b[3], (short)0, acc1, false, false);
    }
#pragma unroll
    for (int r = 0; r < 8; ++r) part[r] += (acc0[r] + acc1[r]) * cav[r];
  }

#pragma unroll
  for (int r = 0; r < 8; ++r) red[lane][r] = part[r];
  __syncthreads();

  if (lane < 16) {
    const int m = lane;
    const int r = m & 7;
    const int l0 = (m < 8) ? 0 : 16;
    float s = 0.0f;
#pragma unroll
    for (int l = 0; l < 16; ++l) s += red[l0 + l][r];
    J[(size_t)(rowBase + m) * DD + o] = sigmoidf_(s);
  }
}

// ---------------------------------------------------------------------------
extern "C" void kernel_launch(void* const* d_in, const int* in_sizes, int n_in,
                              void* d_out, int out_size, void* d_ws, size_t ws_size,
                              hipStream_t stream) {
  const float* img    = (const float*)d_in[0];
  const float* audio  = (const float*)d_in[1];
  const float* vis_W  = (const float*)d_in[2];
  const float* vis_b  = (const float*)d_in[3];
  const float* aud_W  = (const float*)d_in[4];
  const float* aud_b  = (const float*)d_in[5];
  const float* msv_W  = (const float*)d_in[6];
  const float* msv_b  = (const float*)d_in[7];
  const float* msa_W  = (const float*)d_in[8];
  const float* msa_b  = (const float*)d_in[9];
  const float* mmfa_W = (const float*)d_in[10];
  const float* mmfa_b = (const float*)d_in[11];
  const float* bil_W  = (const float*)d_in[12];
  const float* t_c    = (const float*)d_in[13];
  const float* out_W1 = (const float*)d_in[14];
  const float* out_b1 = (const float*)d_in[15];
  const float* out_W2 = (const float*)d_in[16];
  const float* out_b2 = (const float*)d_in[17];
  const float* clsvW1 = (const float*)d_in[18];
  const float* clsvW2 = (const float*)d_in[19];
  const float* clsaW1 = (const float*)d_in[20];
  const float* clsaW2 = (const float*)d_in[21];
  float* outp = (float*)d_out;

  // deterministic bump allocator on d_ws
  size_t off = 0;
  auto allocB = [&](size_t bytes) -> void* {
    off = (off + 255) & ~(size_t)255;
    void* p = (char*)d_ws + off;
    off += bytes;
    return p;
  };
  auto allocH = [&](size_t n) { return (half_t*)allocB(n * sizeof(half_t)); };
  auto allocF = [&](size_t n) { return (float*)allocB(n * sizeof(float)); };

  // f16 weight / input mirrors
  half_t* imgH   = allocH((size_t)CB * IMG);
  half_t* audH   = allocH((size_t)CB * AUD);
  half_t* visWh  = allocH((size_t)ENC * IMG);
  half_t* audWh  = allocH((size_t)ENC * AUD);
  half_t* msvWh  = allocH((size_t)SD * ENC);
  half_t* msaWh  = allocH((size_t)SD * ENC);
  half_t* mmfaWh = allocH((size_t)8 * DD * DD);
  half_t* bilWth = allocH((size_t)DD * DD * DD);
  half_t* outW1h = allocH((size_t)OUT1 * SD);
  half_t* outW2h = allocH((size_t)LAT * OUT1);
  half_t* cv1h   = allocH(32 * LAT);
  half_t* cv2h   = allocH(NCLS * 32);
  half_t* ca1h   = allocH(32 * LAT);
  half_t* ca2h   = allocH(NCLS * 32);

  // activations
  half_t* visH  = allocH((size_t)CB * ENC);
  half_t* audEH = allocH((size_t)CB * ENC);
  float*  vmsF  = allocF((size_t)BS * DD);
  half_t* vmsH  = allocH((size_t)BS * DD);
  float*  amsF  = allocF((size_t)BS * DD);
  half_t* amsH  = allocH((size_t)BS * DD);
  float*  qvF = allocF((size_t)BS * DD);  // reused as L4
  float*  qaF = allocF((size_t)BS * DD);  // reused as L5
  float*  kvF = allocF((size_t)BS * DD);  // reused as L6
  float*  kaF = allocF((size_t)BS * DD);  // reused as L7
  float*  aVAf = allocF((size_t)BS * DD);  half_t* aVAh = allocH((size_t)BS * DD);
  float*  aVf  = allocF((size_t)BS * DD);  half_t* aVh  = allocH((size_t)BS * DD);
  float*  aAVf = allocF((size_t)BS * DD);  half_t* aAVh = allocH((size_t)BS * DD);
  float*  aAf  = allocF((size_t)BS * DD);  half_t* aAh  = allocH((size_t)BS * DD);
  half_t* cvaH = allocH((size_t)BS * DD);
  float*  cavF = allocF((size_t)BS * DD);
  float*  jF   = allocF((size_t)BS * DD);
  float*  MF   = allocF((size_t)BS * DD);   // M1, then reused for M2
  float*  xv2F = allocF((size_t)BS * DD);  half_t* xv2H = allocH((size_t)BS * DD);
  float*  xa2F = allocF((size_t)BS * DD);  half_t* xa2H = allocH((size_t)BS * DD);
  half_t* ovH  = allocH((size_t)BS * DD);
  half_t* oaH  = allocH((size_t)BS * DD);
  half_t* hidVh = allocH((size_t)CB * OUT1);
  half_t* hidAh = allocH((size_t)CB * OUT1);
  half_t* fvH = allocH((size_t)CB * LAT);
  half_t* faH = allocH((size_t)CB * LAT);
  half_t* pvHid = allocH((size_t)CB * 32);
  half_t* paHid = allocH((size_t)CB * 32);

  auto cvt = [&](const float* s, half_t* d, int n) {
    cvt_f32_f16_kernel<<<(n + 255) / 256, 256, 0, stream>>>(s, d, n);
  };
  auto gemm = [&](const half_t* A, const half_t* Bt, const float* bias,
                  float* Cf, half_t* Ch, int M, int N, int K, bool relu) {
    dim3 g((N + 63) / 64, M / 64), b(128);
    if (relu) wmma_gemm_kernel<1><<<g, b, 0, stream>>>(A, Bt, bias, Cf, Ch, M, N, K);
    else      wmma_gemm_kernel<0><<<g, b, 0, stream>>>(A, Bt, bias, Cf, Ch, M, N, K);
  };
  auto emul = [&](const float* a, const float* b, float* oF, half_t* oH, int n) {
    ew_mul_kernel<<<(n + 255) / 256, 256, 0, stream>>>(a, b, oF, oH, n);
  };

  // --- weight / input conversion ---
  cvt(img, imgH, CB * IMG);
  cvt(audio, audH, CB * AUD);
  cvt(vis_W, visWh, ENC * IMG);
  cvt(aud_W, audWh, ENC * AUD);
  cvt(msv_W, msvWh, SD * ENC);
  cvt(msa_W, msaWh, SD * ENC);
  cvt(mmfa_W, mmfaWh, 8 * DD * DD);
  cvt(out_W1, outW1h, OUT1 * SD);
  cvt(out_W2, outW2h, LAT * OUT1);
  cvt(clsvW1, cv1h, 32 * LAT);
  cvt(clsvW2, cv2h, NCLS * 32);
  cvt(clsaW1, ca1h, 32 * LAT);
  cvt(clsaW2, ca2h, NCLS * 32);
  {
    int n = DD * DD * DD;
    bil_transpose_kernel<<<(n + 255) / 256, 256, 0, stream>>>(bil_W, bilWth, n);
  }

  // --- encoders + multiscale ---
  gemm(imgH, visWh, vis_b, nullptr, visH, CB, ENC, IMG, true);
  gemm(audH, audWh, aud_b, nullptr, audEH, CB, ENC, AUD, true);
  gemm(visH, msvWh, msv_b, vmsF, vmsH, CB, SD, ENC, true);
  gemm(audEH, msaWh, msa_b, amsF, amsH, CB, SD, ENC, true);

  // --- MmFA (shared for both calls) ---
  auto mmfa = [&](const half_t* xvH_, const float* xvF_,
                  const half_t* xaH_, const float* xaF_, float* Mout) {
    const int n = BS * DD;
    gemm(xvH_, mmfaWh + 0 * DD * DD, mmfa_b + 0 * DD, qvF, nullptr, BS, DD, DD, false);
    gemm(xaH_, mmfaWh + 1 * DD * DD, mmfa_b + 1 * DD, qaF, nullptr, BS, DD, DD, false);
    gemm(xvH_, mmfaWh + 2 * DD * DD, mmfa_b + 2 * DD, kvF, nullptr, BS, DD, DD, false);
    gemm(xaH_, mmfaWh + 3 * DD * DD, mmfa_b + 3 * DD, kaF, nullptr, BS, DD, DD, false);
    attention_kernel<<<CB, 256, 0, stream>>>(qvF, qaF, kvF, kaF, xvF_, xaF_,
                                             aVAf, aVAh, aVf, aVh, aAVf, aAVh, aAf, aAh);
    // L4..L7 overwrite q/k buffers (already consumed by attention)
    gemm(aVAh, mmfaWh + 4 * DD * DD, mmfa_b + 4 * DD, qvF, nullptr, BS, DD, DD, false);
    gemm(aVh,  mmfaWh + 5 * DD * DD, mmfa_b + 5 * DD, qaF, nullptr, BS, DD, DD, false);
    gemm(aAVh, mmfaWh + 6 * DD * DD, mmfa_b + 6 * DD, kvF, nullptr, BS, DD, DD, false);
    gemm(aAh,  mmfaWh + 7 * DD * DD, mmfa_b + 7 * DD, kaF, nullptr, BS, DD, DD, false);
    gate_kernel<<<(n + 255) / 256, 256, 0, stream>>>(qvF, qaF, aVf, kvF, kaF, aAf,
                                                     cvaH, cavF, n);
    bilinear_kernel<<<dim3(BS / 16, DD), 32, 0, stream>>>(cvaH, cavF, bilWth, jF);
    combine_kernel<<<(n + 255) / 256, 256, 0, stream>>>(jF, xvF_, xaF_, t_c, Mout, n);
  };

  mmfa(vmsH, vmsF, amsH, amsF, MF);                 // M1
  emul(MF, vmsF, xv2F, xv2H, BS * DD);              // x_v2 = M1 * v_ms
  emul(MF, amsF, xa2F, xa2H, BS * DD);              // x_a2 = M1 * a_ms
  mmfa(xv2H, xv2F, xa2H, xa2F, MF);                 // M2 (reuses MF)
  emul(MF, vmsF, nullptr, ovH, BS * DD);            // ov = M2 * v_ms  [B,768]
  emul(MF, amsF, nullptr, oaH, BS * DD);            // oa = M2 * a_ms

  // --- out_layer + classifiers ---
  gemm(ovH, outW1h, out_b1, nullptr, hidVh, CB, OUT1, SD, true);
  gemm(hidVh, outW2h, out_b2, outp, fvH, CB, LAT, OUT1, false);               // fv
  gemm(oaH, outW1h, out_b1, nullptr, hidAh, CB, OUT1, SD, true);
  gemm(hidAh, outW2h, out_b2, outp + (size_t)CB * LAT, faH, CB, LAT, OUT1, false); // fa

  gemm(fvH, cv1h, nullptr, nullptr, pvHid, CB, 32, LAT, true);
  gemm(pvHid, cv2h, nullptr, outp + 2 * (size_t)CB * LAT, nullptr, CB, NCLS, 32, false); // pv
  gemm(faH, ca1h, nullptr, nullptr, paHid, CB, 32, LAT, true);
  gemm(paHid, ca2h, nullptr, outp + 2 * (size_t)CB * LAT + (size_t)CB * NCLS,
       nullptr, CB, NCLS, 32, false);                                         // pa
}
